// GATExt_10024453669352
// MI455X (gfx1250) — compile-verified, matrix operand order checked by reading, not run on hardware
//
#include <hip/hip_runtime.h>
#include <hip/hip_bf16.h>
#include <math.h>

#define N_NODES  50000
#define N_EDGES0 400000
#define N_EDGES  (N_EDGES0 + N_NODES)   // reference appends self loops

typedef __attribute__((ext_vector_type(16))) __bf16 v16bf;
typedef __attribute__((ext_vector_type(8)))  __bf16 v8bf;
typedef __attribute__((ext_vector_type(8)))  float  v8f;

// ---------------------------------------------------------------- utilities
__global__ void fill_f32(float* __restrict__ p, float v, int n) {
  int i = blockIdx.x * blockDim.x + threadIdx.x;
  if (i < n) p[i] = v;
}

// W[fin][HC] fp32 -> Bt[HC][fin] bf16 (transposed so B columns are contiguous)
__global__ void convert_wt(const float* __restrict__ W, __bf16* __restrict__ Bt,
                           int K, int Nc) {
  int i = blockIdx.x * blockDim.x + threadIdx.x;
  if (i >= K * Nc) return;
  int k = i / Nc, c = i - k * Nc;
  Bt[(size_t)c * K + k] = (__bf16)W[i];
}

// float atomic max: signed-int max for positives, unsigned-int min for negatives
__device__ __forceinline__ void atomicMaxF(float* addr, float val) {
  if (val >= 0.0f) atomicMax((int*)addr, __float_as_int(val));
  else             atomicMin((unsigned int*)addr, __float_as_uint(val));
}

// ---------------------------------------------------------------- WMMA GEMM
// One wave computes a 16 x (16*NT) tile of BOTH  xl = A@Wl  and  xr = A@Wr.
// A fp32 [M][K] converted to bf16 in-register; Bt* are bf16 [Nc][K] (transposed).
template<int NT>
__global__ void __launch_bounds__(256)
gemm_dual_wmma(const float*  __restrict__ A,
               const __bf16* __restrict__ BtL,
               const __bf16* __restrict__ BtR,
               float* __restrict__ CL, float* __restrict__ CR,
               int M, int K, int Nc) {
  const int lane = threadIdx.x & 31;
  const int wave = (int)((blockIdx.x * blockDim.x + threadIdx.x) >> 5);
  const int ntg  = Nc / (16 * NT);
  const int mt   = wave / ntg;
  const int ng   = wave - mt * ntg;
  if (mt * 16 >= M) return;                 // whole-wave uniform exit
  const int r    = lane & 15;
  const int half = lane >> 4;

  const float* Arow = A + (size_t)(mt * 16 + r) * K;

  v8f accL[NT] = {};
  v8f accR[NT] = {};

  for (int kb = 0; kb < K; kb += 32) {
    // A fragment: lane holds row r, K = {kb+half*8 .. +7} U {kb+16+half*8 .. +7}
    const float* ap = Arow + kb + (half << 3);
    float4 f0 = *(const float4*)(ap);
    float4 f1 = *(const float4*)(ap + 4);
    float4 f2 = *(const float4*)(ap + 16);
    float4 f3 = *(const float4*)(ap + 20);
    v16bf a;
    a[0]=(__bf16)f0.x; a[1]=(__bf16)f0.y; a[2]=(__bf16)f0.z; a[3]=(__bf16)f0.w;
    a[4]=(__bf16)f1.x; a[5]=(__bf16)f1.y; a[6]=(__bf16)f1.z; a[7]=(__bf16)f1.w;
    a[8]=(__bf16)f2.x; a[9]=(__bf16)f2.y; a[10]=(__bf16)f2.z; a[11]=(__bf16)f2.w;
    a[12]=(__bf16)f3.x; a[13]=(__bf16)f3.y; a[14]=(__bf16)f3.z; a[15]=(__bf16)f3.w;

#pragma unroll
    for (int t = 0; t < NT; t++) {
      const int ncol = (ng * NT + t) * 16 + r;
      {
        const __bf16* bp = BtL + (size_t)ncol * K + kb + (half << 3);
        v8bf blo = *(const v8bf*)bp;
        v8bf bhi = *(const v8bf*)(bp + 16);
        v16bf b;
#pragma unroll
        for (int j = 0; j < 8; j++) { b[j] = blo[j]; b[j + 8] = bhi[j]; }
        accL[t] = __builtin_amdgcn_wmma_f32_16x16x32_bf16(
            false, a, false, b, (short)0, accL[t], false, false);
      }
      {
        const __bf16* bp = BtR + (size_t)ncol * K + kb + (half << 3);
        v8bf blo = *(const v8bf*)bp;
        v8bf bhi = *(const v8bf*)(bp + 16);
        v16bf b;
#pragma unroll
        for (int j = 0; j < 8; j++) { b[j] = blo[j]; b[j + 8] = bhi[j]; }
        accR[t] = __builtin_amdgcn_wmma_f32_16x16x32_bf16(
            false, a, false, b, (short)0, accR[t], false, false);
      }
    }
  }

  // C/D layout: VGPR v -> row (v + half*8), col = r within the 16x16 tile
#pragma unroll
  for (int t = 0; t < NT; t++) {
    const int ncol = (ng * NT + t) * 16 + r;
#pragma unroll
    for (int v = 0; v < 8; v++) {
      size_t row = (size_t)(mt * 16 + half * 8 + v);
      CL[row * Nc + ncol] = accL[t][v];
      CR[row * Nc + ncol] = accR[t][v];
    }
  }
}

// ---------------------------------------------------------------- edge phase
// Wave per edge: logits[e][h] = sum_c att[h][c]*leakyrelu(xl[src]+xr[dst]); max into m[dst]
__global__ void edge_logits(const int* __restrict__ ei,
                            const float* __restrict__ xl,
                            const float* __restrict__ xr,
                            const float* __restrict__ att,
                            float* __restrict__ logits,
                            float* __restrict__ mmax,
                            int H, int C) {
  int e = (int)((blockIdx.x * blockDim.x + threadIdx.x) >> 5);
  int lane = threadIdx.x & 31;
  if (e >= N_EDGES) return;
  int src, dst;
  if (e < N_EDGES0) { src = ei[e]; dst = ei[N_EDGES0 + e]; }
  else              { src = dst = e - N_EDGES0; }          // self loop
  const int HC = H * C;
  for (int h = 0; h < H; h++) {
    const float* pl = xl + (size_t)src * HC + h * C;
    const float* pr = xr + (size_t)dst * HC + h * C;
    const float* pa = att + h * C;
    float s = 0.0f;
    for (int c = lane; c < C; c += 32) {
      float v = pl[c] + pr[c];
      v = (v > 0.0f) ? v : 0.2f * v;                       // leaky relu 0.2
      s += v * pa[c];
    }
#pragma unroll
    for (int off = 16; off > 0; off >>= 1) s += __shfl_down(s, off, 32);
    if (lane == 0) {
      logits[(size_t)e * H + h] = s;
      atomicMaxF(&mmax[dst * H + h], s);
    }
  }
}

// Thread per (edge, head): p = exp(logit - m[dst]); denom[dst] += p
__global__ void edge_softmax_num(const int* __restrict__ ei,
                                 float* __restrict__ logits,
                                 const float* __restrict__ mmax,
                                 float* __restrict__ denom, int H) {
  int i = blockIdx.x * blockDim.x + threadIdx.x;
  if (i >= N_EDGES * H) return;
  int e = i / H, h = i - e * H;
  int dst = (e < N_EDGES0) ? ei[N_EDGES0 + e] : (e - N_EDGES0);
  float p = __expf(logits[i] - mmax[dst * H + h]);
  logits[i] = p;
  atomicAdd(&denom[dst * H + h], p);
}

// Wave per edge: out[dst] += (p/denom[dst]) * xl[src]
__global__ void edge_aggregate(const int* __restrict__ ei,
                               const float* __restrict__ xl,
                               const float* __restrict__ logits,
                               const float* __restrict__ denom,
                               float* __restrict__ out,
                               int H, int C) {
  int e = (int)((blockIdx.x * blockDim.x + threadIdx.x) >> 5);
  int lane = threadIdx.x & 31;
  if (e >= N_EDGES) return;
  int src, dst;
  if (e < N_EDGES0) { src = ei[e]; dst = ei[N_EDGES0 + e]; }
  else              { src = dst = e - N_EDGES0; }
  const int HC = H * C;
  for (int h = 0; h < H; h++) {
    float alpha = logits[(size_t)e * H + h] / denom[dst * H + h];
    const float* pl = xl + (size_t)src * HC + h * C;
    float*       po = out + (size_t)dst * HC + h * C;
    for (int c = lane; c < C; c += 32)
      atomicAdd(&po[c], alpha * pl[c]);
  }
}

__global__ void add_bias(float* __restrict__ out, const float* __restrict__ bias, int HC) {
  int i = blockIdx.x * blockDim.x + threadIdx.x;
  if (i >= N_NODES * HC) return;
  out[i] += bias[i & (HC - 1)];   // HC is 256 or 32 (power of two)
}

// ---------------------------------------------------------------- launcher
extern "C" void kernel_launch(void* const* d_in, const int* in_sizes, int n_in,
                              void* d_out, int out_size, void* d_ws, size_t ws_size,
                              hipStream_t stream) {
  (void)in_sizes; (void)n_in; (void)out_size; (void)ws_size;
  const float* x  = (const float*)d_in[0];
  const int*   ei = (const int*)d_in[1];

  // workspace carving (256B aligned)
  size_t off = 0;
  auto take = [&](size_t bytes) -> char* {
    char* p = (char*)d_ws + off;
    off += (bytes + 255) & ~(size_t)255;
    return p;
  };
  float*  P0     = (float*)take((size_t)N_NODES * 256 * sizeof(float));
  float*  P1     = (float*)take((size_t)N_NODES * 256 * sizeof(float));
  float*  xl     = (float*)take((size_t)N_NODES * 256 * sizeof(float));
  float*  xr     = (float*)take((size_t)N_NODES * 256 * sizeof(float));
  float*  logits = (float*)take((size_t)N_EDGES * 2 * sizeof(float));
  float*  mmax   = (float*)take((size_t)N_NODES * 2 * sizeof(float));
  float*  denom  = (float*)take((size_t)N_NODES * 2 * sizeof(float));
  __bf16* wlt    = (__bf16*)take((size_t)1024 * 256 * sizeof(__bf16));
  __bf16* wrt    = (__bf16*)take((size_t)1024 * 256 * sizeof(__bf16));

  const int fins[4] = {1024, 256, 256, 256};
  const int Hs[4]   = {2, 2, 2, 1};
  const int Cs[4]   = {128, 128, 128, 32};
  float* bufs[2] = {P0, P1};

  const float* hin = x;
  for (int li = 0; li < 4; li++) {
    const int fin = fins[li], H = Hs[li], C = Cs[li], HC = H * C;
    const float* Wl   = (const float*)d_in[2 + 4 * li];
    const float* Wr   = (const float*)d_in[3 + 4 * li];
    const float* att  = (const float*)d_in[4 + 4 * li];
    const float* bias = (const float*)d_in[5 + 4 * li];
    float* hout = (li == 3) ? (float*)d_out : bufs[li & 1];

    // 1) weight transpose + bf16 convert
    const int wn = fin * HC;
    convert_wt<<<(wn + 255) / 256, 256, 0, stream>>>(Wl, wlt, fin, HC);
    convert_wt<<<(wn + 255) / 256, 256, 0, stream>>>(Wr, wrt, fin, HC);

    // 2) dual WMMA GEMM: xl = h@Wl, xr = h@Wr
    if ((HC % 64) == 0) {
      int waves = (N_NODES / 16) * (HC / 64);
      gemm_dual_wmma<4><<<(waves * 32 + 255) / 256, 256, 0, stream>>>(
          hin, wlt, wrt, xl, xr, N_NODES, fin, HC);
    } else {
      int waves = (N_NODES / 16) * (HC / 32);
      gemm_dual_wmma<2><<<(waves * 32 + 255) / 256, 256, 0, stream>>>(
          hin, wlt, wrt, xl, xr, N_NODES, fin, HC);
    }

    // 3) init accumulators
    fill_f32<<<(N_NODES * HC + 255) / 256, 256, 0, stream>>>(hout, 0.0f, N_NODES * HC);
    fill_f32<<<(N_NODES * H + 255) / 256, 256, 0, stream>>>(mmax, -INFINITY, N_NODES * H);
    fill_f32<<<(N_NODES * H + 255) / 256, 256, 0, stream>>>(denom, 0.0f, N_NODES * H);

    // 4) edge phase (segment softmax + scatter)
    const int ethreads = N_EDGES * 32;
    edge_logits<<<(ethreads + 255) / 256, 256, 0, stream>>>(ei, xl, xr, att, logits, mmax, H, C);
    edge_softmax_num<<<(N_EDGES * H + 255) / 256, 256, 0, stream>>>(ei, logits, mmax, denom, H);
    edge_aggregate<<<(ethreads + 255) / 256, 256, 0, stream>>>(ei, xl, logits, denom, hout, H, C);

    // 5) bias
    add_bias<<<(N_NODES * HC + 255) / 256, 256, 0, stream>>>(hout, bias, HC);

    hin = hout;
  }
}
